// VectorQuantizerEMA_43233140802218
// MI455X (gfx1250) — matrix operand order repaired and checked by reading, not compiled, not run on hardware
//
#include <hip/hip_runtime.h>
#include <hip/hip_bf16.h>
#include <math.h>

// ---------------- problem constants ----------------
#define NUM_EMB   1024
#define EMB_DIM   256
#define N_VEC     32768          // 32 * 32 * 32 vectors
#define TM        128            // rows (vectors) per block in argmin kernel
#define NBLK      (N_VEC / TM)   // 256 blocks
#define ARS       264            // padded LDS row stride in ushorts (256 + 8)
#define BROWS     (16 * ARS)     // ushorts per 16-codeword B panel

// d_out layout (floats): [loss][quantized BCHW 8388608][perplexity][encodings 33554432]
#define OUT_LOSS  0
#define OUT_Q     1
#define QELEMS    8388608
#define OUT_PERP  8388609
#define OUT_ENC   8388610
#define ENC_ELEMS 33554432

// ---------------- WMMA types ----------------
typedef __attribute__((ext_vector_type(16))) __bf16 v16bf;
typedef __attribute__((ext_vector_type(8)))  float  v8f;

union BFVec { v16bf v; float4 f4[2]; };

__device__ inline v8f wmma_bf16(v16bf a, v16bf b, v8f c) {
  // D = A(16x32 bf16) * B(32x16 bf16) + C(16x16 f32)
  return __builtin_amdgcn_wmma_f32_16x16x32_bf16(
      /*neg_a=*/false, a, /*neg_b=*/false, b,
      /*c_mod=*/(short)0, c, /*reuse_a=*/false, /*reuse_b=*/false);
}

__device__ inline unsigned short f32_to_bf16_rne(float f) {
  unsigned int u = __float_as_uint(f);
  u += 0x7fffu + ((u >> 16) & 1u);     // round to nearest even
  return (unsigned short)(u >> 16);
}
__device__ inline float bf16_to_f32(unsigned short h) {
  return __uint_as_float(((unsigned int)h) << 16);
}

// ---------------- kernel 0: zero encodings + counters ----------------
__global__ void vq_zero_kernel(float* __restrict__ out, int* __restrict__ counts,
                               float* __restrict__ sse) {
  const size_t tid    = (size_t)blockIdx.x * blockDim.x + threadIdx.x;
  const size_t stride = (size_t)gridDim.x * blockDim.x;
  float2* e2 = (float2*)(out + OUT_ENC);          // 8-byte aligned region
  const size_t n2 = ENC_ELEMS / 2;
  const float2 z = make_float2(0.f, 0.f);
  for (size_t i = tid; i < n2; i += stride) e2[i] = z;
  if (blockIdx.x == 0) {
    for (int k = threadIdx.x; k < NUM_EMB; k += blockDim.x) counts[k] = 0;
    if (threadIdx.x == 0) *sse = 0.f;
  }
}

// ---------------- kernel 1: codebook -> bf16 hi/lo split + ||e||^2 ----------------
__global__ void vq_prep_kernel(const float* __restrict__ emb,
                               unsigned short* __restrict__ ehi,
                               unsigned short* __restrict__ elo,
                               float* __restrict__ enorm) {
  const int k = blockIdx.x;        // codeword
  const int c = threadIdx.x;       // dim
  const float v = emb[((size_t)k << 8) + c];
  const unsigned short h = f32_to_bf16_rne(v);
  ehi[((size_t)k << 8) + c] = h;
  elo[((size_t)k << 8) + c] = f32_to_bf16_rne(v - bf16_to_f32(h));

  float s = v * v;
  #pragma unroll
  for (int off = 16; off > 0; off >>= 1) s += __shfl_down(s, off, 32);
  __shared__ float red[8];
  if ((c & 31) == 0) red[c >> 5] = s;
  __syncthreads();
  if (c == 0) {
    float t = 0.f;
    #pragma unroll
    for (int w = 0; w < 8; ++w) t += red[w];
    enorm[k] = t;
  }
}

// ---------------- kernel 2: split-bf16 WMMA GEMM + argmin ----------------
__global__ void __launch_bounds__(256)
vq_argmin_kernel(const float* __restrict__ x,
                 const unsigned short* __restrict__ ehi,
                 const unsigned short* __restrict__ elo,
                 const float* __restrict__ enorm,
                 int* __restrict__ out_idx,
                 float* __restrict__ enc,
                 int* __restrict__ counts) {
  extern __shared__ unsigned char smem[];
  unsigned short* Ahi  = (unsigned short*)smem;         // [TM][ARS]
  unsigned short* Alo  = Ahi + TM * ARS;
  unsigned short* Bbuf = Alo + TM * ARS;                // 2 x (hi,lo) x [16][ARS]
  float*          En   = (float*)(Bbuf + 4 * BROWS);    // [NUM_EMB]

  const int tid  = threadIdx.x;
  const int lane = tid & 31;
  const int wave = tid >> 5;                // 0..7 (M-tiles)
  const int blk  = blockIdx.x;              // 0..255
  const int b    = blk >> 3;                // batch index
  const int hw0  = (blk & 7) << 7;          // 128 hw positions per block

  // ---- stage x tile (BCHW strided -> LDS row-major bf16 hi/lo in A layout) ----
  for (int i = tid; i < (TM * EMB_DIM) / 4; i += 256) {
    const int c  = i >> 5;                  // channel 0..255
    const int j4 = (i & 31) << 2;           // local row 0,4,...,124
    const float4 v4 = *(const float4*)(x + (((size_t)(b << 8) + c) << 10) + hw0 + j4);
    const float vv[4] = {v4.x, v4.y, v4.z, v4.w};
    #pragma unroll
    for (int t = 0; t < 4; ++t) {
      const unsigned short h = f32_to_bf16_rne(vv[t]);
      Ahi[(j4 + t) * ARS + c] = h;
      Alo[(j4 + t) * ARS + c] = f32_to_bf16_rne(vv[t] - bf16_to_f32(h));
    }
  }
  for (int i = tid; i < NUM_EMB; i += 256) En[i] = enorm[i];

  // B-panel staging geometry: 512 uint4 per panel, 2 per thread per (hi|lo)
  int ur[2], uc[2];
  #pragma unroll
  for (int q = 0; q < 2; ++q) {
    const int idx = (tid << 1) + q;
    ur[q] = idx >> 5;                       // row 0..15
    uc[q] = (idx & 31) << 3;                // ushort col, multiple of 8 (16B)
  }

  // ---- prologue: stage B chunk 0 into buffer 0 ----
  {
    uint4 sh[2], sl[2];
    #pragma unroll
    for (int q = 0; q < 2; ++q) {
      const size_t g = (((size_t)ur[q]) << 8) + uc[q];
      sh[q] = *(const uint4*)(ehi + g);
      sl[q] = *(const uint4*)(elo + g);
    }
    #pragma unroll
    for (int q = 0; q < 2; ++q) {
      const int o = ur[q] * ARS + uc[q];
      *(uint4*)(Bbuf + o) = sh[q];
      *(uint4*)(Bbuf + BROWS + o) = sl[q];
    }
  }
  __syncthreads();

  // ---- hoist this wave's A operands (16 rows x 256 K) into VGPRs ----
  // A-layout (ISA 7.12.2): lane m holds row M=m%16; lanes 0-15 get K{0..7,16..23},
  // lanes 16-31 get K{8..15,24..31} of each 32-K chunk.
  v16bf Ah[8], Al[8];
  {
    const int arow = (wave << 4) + (lane & 15);
    const unsigned short* ah = Ahi + arow * ARS + ((lane >> 4) << 3); // +16B upper half
    const unsigned short* al = Alo + arow * ARS + ((lane >> 4) << 3);
    #pragma unroll
    for (int kc = 0; kc < 8; ++kc) {
      BFVec t;
      t.f4[0] = *(const float4*)(ah + kc * 32);
      t.f4[1] = *(const float4*)(ah + kc * 32 + 16);
      Ah[kc] = t.v;
      t.f4[0] = *(const float4*)(al + kc * 32);
      t.f4[1] = *(const float4*)(al + kc * 32 + 16);
      Al[kc] = t.v;
    }
  }

  float minv[8];
  int   mini[8];
  #pragma unroll
  for (int r = 0; r < 8; ++r) { minv[r] = 3.0e38f; mini[r] = 0; }

  const int bcol = lane & 15;               // column within 16-wide N chunk

  for (int nc = 0; nc < NUM_EMB / 16; ++nc) {
    const int ncn = nc + 1;

    // ---- issue next chunk's global loads into registers (overlaps WMMA below) ----
    uint4 nh[2], nl[2];
    if (ncn < NUM_EMB / 16) {
      #pragma unroll
      for (int q = 0; q < 2; ++q) {
        const size_t g = (((size_t)((ncn << 4) + ur[q])) << 8) + uc[q];
        nh[q] = *(const uint4*)(ehi + g);
        nl[q] = *(const uint4*)(elo + g);
      }
    }

    // ---- 3-term split-bf16 accumulation: xh*eh + xh*el + xl*eh ----
    const unsigned short* Bh = Bbuf + (nc & 1) * (2 * BROWS);
    const unsigned short* Bl = Bh + BROWS;
    v8f acc = {0.f, 0.f, 0.f, 0.f, 0.f, 0.f, 0.f, 0.f};
    // B-layout: lane n holds col N=n%16; lanes 0-15 K{0..7,16..23}, lanes 16-31 K{8..15,24..31}.
    const unsigned short* bh0 = Bh + bcol * ARS + ((lane >> 4) << 3); // +16B upper
    const unsigned short* bl0 = Bl + bcol * ARS + ((lane >> 4) << 3);
    #pragma unroll
    for (int kc = 0; kc < 8; ++kc) {
      BFVec bh, bl;
      bh.f4[0] = *(const float4*)(bh0 + kc * 32);
      bh.f4[1] = *(const float4*)(bh0 + kc * 32 + 16);
      bl.f4[0] = *(const float4*)(bl0 + kc * 32);
      bl.f4[1] = *(const float4*)(bl0 + kc * 32 + 16);
      acc = wmma_bf16(Ah[kc], bh.v, acc);
      acc = wmma_bf16(Ah[kc], bl.v, acc);
      acc = wmma_bf16(Al[kc], bh.v, acc);
    }

    // dist(m, n) - ||x||^2 = ||e||^2 - 2 x.e  (row-constant term dropped)
    const int   ncol = (nc << 4) + bcol;
    const float en   = En[ncol];
    #pragma unroll
    for (int r = 0; r < 8; ++r) {
      const float d = fmaf(-2.0f, acc[r], en);
      if (d < minv[r]) { minv[r] = d; mini[r] = ncol; }  // strict < keeps first idx
    }

    // ---- commit next chunk to the alternate LDS buffer ----
    if (ncn < NUM_EMB / 16) {
      unsigned short* Bhn = Bbuf + (ncn & 1) * (2 * BROWS);
      #pragma unroll
      for (int q = 0; q < 2; ++q) {
        const int o = ur[q] * ARS + uc[q];
        *(uint4*)(Bhn + o) = nh[q];
        *(uint4*)(Bhn + BROWS + o) = nl[q];
      }
    }
    __syncthreads();
  }

  // ---- per-row argmin across 16 lanes of each half-wave (first-index tie-break) ----
  #pragma unroll
  for (int r = 0; r < 8; ++r) {
    #pragma unroll
    for (int off = 8; off > 0; off >>= 1) {
      const float ov = __shfl_xor(minv[r], off, 16);
      const int   oi = __shfl_xor(mini[r], off, 16);
      if (ov < minv[r] || (ov == minv[r] && oi < mini[r])) { minv[r] = ov; mini[r] = oi; }
    }
  }
  // C/D layout: VGPR r -> row r (lanes 0-15) / row r+8 (lanes 16-31)
  if ((lane & 15) == 0) {
    const int rowbase = blk * TM + (wave << 4) + ((lane >> 4) << 3);
    #pragma unroll
    for (int r = 0; r < 8; ++r) {
      const int row = rowbase + r;
      const int k   = mini[r];
      out_idx[row] = k;
      enc[((size_t)row << 10) + k] = 1.0f;     // one-hot
      atomicAdd(&counts[k], 1);
    }
  }
}

// ---------------- kernel 3: gather quantized (BCHW) + SSE ----------------
__global__ void vq_gather_kernel(const float* __restrict__ x,
                                 const float* __restrict__ emb,
                                 const int* __restrict__ idx,
                                 float* __restrict__ qout,
                                 float* __restrict__ sse) {
  const int bid = blockIdx.x;           // 0..8191 = (b, c)
  const int b   = bid >> 8;
  const int c   = bid & 255;
  const int tid = threadIdx.x;
  const float* xrow = x    + (((size_t)(b << 8) + c) << 10);
  float*       qrow = qout + (((size_t)(b << 8) + c) << 10);
  const int*   rows = idx + (b << 10);
  float local = 0.f;
  #pragma unroll
  for (int t = 0; t < 4; ++t) {
    const int hw = tid + (t << 8);
    const int k  = rows[hw];
    const float q  = emb[((size_t)k << 8) + c];
    const float xv = xrow[hw];
    qrow[hw] = q;                       // x + stop_grad(q - x) == q in value
    const float d = q - xv;
    local = fmaf(d, d, local);
  }
  #pragma unroll
  for (int off = 16; off > 0; off >>= 1) local += __shfl_down(local, off, 32);
  __shared__ float red[8];
  if ((tid & 31) == 0) red[tid >> 5] = local;
  __syncthreads();
  if (tid == 0) {
    float t = 0.f;
    #pragma unroll
    for (int w = 0; w < 8; ++w) t += red[w];
    atomicAdd(sse, t);
  }
}

// ---------------- kernel 4: loss + perplexity ----------------
__global__ void vq_finalize_kernel(const int* __restrict__ counts,
                                   const float* __restrict__ sse,
                                   float* __restrict__ out) {
  const int tid = threadIdx.x;
  float h = 0.f;
  for (int k = tid; k < NUM_EMB; k += 256) {
    const float p = (float)counts[k] * (1.0f / (float)N_VEC);
    h -= p * logf(p + 1e-10f);
  }
  #pragma unroll
  for (int off = 16; off > 0; off >>= 1) h += __shfl_down(h, off, 32);
  __shared__ float red[8];
  if ((tid & 31) == 0) red[tid >> 5] = h;
  __syncthreads();
  if (tid == 0) {
    float H = 0.f;
    for (int w = 0; w < 8; ++w) H += red[w];
    out[OUT_PERP] = expf(H);
    out[OUT_LOSS] = 0.25f * (*sse) * (1.0f / ((float)N_VEC * (float)EMB_DIM));
  }
}

// ---------------- host entry ----------------
extern "C" void kernel_launch(void* const* d_in, const int* in_sizes, int n_in,
                              void* d_out, int out_size, void* d_ws, size_t ws_size,
                              hipStream_t stream) {
  (void)in_sizes; (void)n_in; (void)out_size; (void)ws_size;
  const float* x   = (const float*)d_in[0];   // (32,256,32,32) f32
  const float* emb = (const float*)d_in[1];   // (1024,256) f32
  float* out = (float*)d_out;

  char* ws = (char*)d_ws;
  unsigned short* ehi   = (unsigned short*)(ws);             // 512 KB
  unsigned short* elo   = (unsigned short*)(ws + 524288);    // 512 KB
  float*          enorm = (float*)(ws + 1048576);            // 4 KB
  int*            cnts  = (int*)(ws + 1052672);              // 4 KB
  int*            inds  = (int*)(ws + 1056768);              // 128 KB
  float*          sse   = (float*)(ws + 1187840);            // 4 B

  vq_zero_kernel<<<4096, 256, 0, stream>>>(out, cnts, sse);
  vq_prep_kernel<<<NUM_EMB, 256, 0, stream>>>(emb, ehi, elo, enorm);

  const size_t smem = (size_t)(2 * TM * ARS + 4 * BROWS) * sizeof(unsigned short)
                      + NUM_EMB * sizeof(float);             // 173,056 B dynamic LDS
  vq_argmin_kernel<<<NBLK, 256, smem, stream>>>(x, ehi, elo, enorm, inds,
                                                out + OUT_ENC, cnts);

  vq_gather_kernel<<<32 * 256, 256, 0, stream>>>(x, emb, inds, out + OUT_Q, sse);
  vq_finalize_kernel<<<1, 256, 0, stream>>>(cnts, sse, out);
}
